// MaskRCNNBoxProposal_17892833755383
// MI455X (gfx1250) — matrix operand order repaired and checked by reading, not compiled, not run on hardware
//
#include <hip/hip_runtime.h>
#include <hip/hip_fp16.h>
#include <math.h>

// ---------------- constants matching the reference ----------------
#define TOT     159882          // anchors per image (sum over 5 levels)
#define KTOT    4507            // top-k candidates per image (1000*4 + 507)
#define NPOST   1000
#define BATCH   2
#define NCH     256
#define BBOX_CLIP 4.135166556742356f   // log(1000/16)

__device__ __constant__ int   c_H[5]    = {200,100,50,25,13};
__device__ __constant__ int   c_W[5]    = {200,100,50,25,13};
__device__ __constant__ float c_STR[5]  = {4.f,8.f,16.f,32.f,61.f};   // IMG//gh
__device__ __constant__ float c_SZ[5]   = {32.f,64.f,128.f,256.f,512.f};
__device__ __constant__ int   c_OFF[6]  = {0,120000,150000,157500,159375,159882};
__device__ __constant__ int   c_KOFF[5] = {0,1000,2000,3000,4000};
__device__ __constant__ int   c_FH[4]   = {200,100,50,25};
__device__ __constant__ float c_SC[4]   = {0.25f,0.125f,0.0625f,0.03125f};

struct Ptr5 { const float* p[5]; };
struct Ptr4 { const float* p[4]; };

typedef __attribute__((ext_vector_type(16))) _Float16 v16h;
typedef __attribute__((ext_vector_type(8)))  _Float16 v8h;
typedef __attribute__((ext_vector_type(8)))  float    v8f;

// monotonic float->uint key (ascending)
static __device__ __forceinline__ unsigned fkey(float f) {
    unsigned u = __float_as_uint(f);
    return u ^ ((u >> 31) ? 0xFFFFFFFFu : 0x80000000u);
}

// ---------------- kernel 1: anchors + decode + clip ----------------
__global__ void decode_kernel(Ptr5 obj, Ptr5 dl, float* __restrict__ prop,
                              float* __restrict__ raw) {
    int gid = blockIdx.x * blockDim.x + threadIdx.x;
    if (gid >= BATCH * TOT) return;
    int b = gid / TOT, j = gid % TOT;
    int l = 0;
    while (l < 4 && j >= c_OFF[l + 1]) ++l;
    int jj = j - c_OFF[l];
    int a = jj % 3, cell = jj / 3;
    int W = c_W[l], H = c_H[l];
    int x = cell % W, y = cell / W;

    float r  = (a == 0) ? 0.5f : (a == 1 ? 1.0f : 2.0f);
    float hr = sqrtf(r);
    float wsz = c_SZ[l] / hr, hsz = c_SZ[l] * hr;
    // jnp.round == round-half-even == rintf
    float cx0 = rintf(-0.5f * wsz), cy0 = rintf(-0.5f * hsz);
    float cx1 = rintf( 0.5f * wsz), cy1 = rintf( 0.5f * hsz);
    float sx = x * c_STR[l], sy = y * c_STR[l];
    float ax1 = sx + cx0, ay1 = sy + cy0, ax2 = sx + cx1, ay2 = sy + cy1;
    float aw = ax2 - ax1, ah = ay2 - ay1;
    float acx = ax1 + 0.5f * aw, acy = ay1 + 0.5f * ah;

    const float* D = dl.p[l];
    const float* O = obj.p[l];
    int hw = H * W, pix = y * W + x;
    float dx = D[(b * 12 + a * 4 + 0) * hw + pix];
    float dy = D[(b * 12 + a * 4 + 1) * hw + pix];
    float dw = fminf(D[(b * 12 + a * 4 + 2) * hw + pix], BBOX_CLIP);
    float dh = fminf(D[(b * 12 + a * 4 + 3) * hw + pix], BBOX_CLIP);

    float pcx = dx * aw + acx, pcy = dy * ah + acy;
    float pw = expf(dw) * aw,  ph = expf(dh) * ah;
    float bx1 = pcx - 0.5f * pw, by1 = pcy - 0.5f * ph;
    float bx2 = pcx + 0.5f * pw, by2 = pcy + 0.5f * ph;
    bx1 = fminf(fmaxf(bx1, 0.f), 800.f);
    by1 = fminf(fmaxf(by1, 0.f), 800.f);
    bx2 = fminf(fmaxf(bx2, 0.f), 800.f);
    by2 = fminf(fmaxf(by2, 0.f), 800.f);

    float* pr = prop + (size_t)(b * TOT + j) * 4;
    pr[0] = bx1; pr[1] = by1; pr[2] = bx2; pr[3] = by2;
    raw[b * TOT + j] = O[(b * 3 + a) * hw + pix];
}

// ------- kernel 2: exact per-(b,level) top-k via 4-round radix select -------
__global__ void topk_kernel(const float* __restrict__ raw,
                            int* __restrict__ candIdx, float* __restrict__ candSc) {
    int b = blockIdx.x / 5, l = blockIdx.x % 5;
    int n = c_OFF[l + 1] - c_OFF[l];
    int k = n < NPOST ? n : NPOST;
    const float* s = raw + b * TOT + c_OFF[l];

    __shared__ unsigned hist[256];
    __shared__ unsigned sh_prefix;
    __shared__ int sh_remaining;
    __shared__ int cntG, cntT;
    if (threadIdx.x == 0) { sh_prefix = 0u; sh_remaining = k; }
    __syncthreads();

    for (int rd = 0; rd < 4; ++rd) {
        int shift = 24 - 8 * rd;
        for (int i = threadIdx.x; i < 256; i += blockDim.x) hist[i] = 0u;
        __syncthreads();
        unsigned pfx = sh_prefix;
        unsigned maskHi = (rd == 0) ? 0u : (0xFFFFFFFFu << (32 - 8 * rd));
        for (int j = threadIdx.x; j < n; j += blockDim.x) {
            unsigned key = fkey(s[j]);
            if ((key & maskHi) == pfx) atomicAdd(&hist[(key >> shift) & 255u], 1u);
        }
        __syncthreads();
        if (threadIdx.x == 0) {
            int rem = sh_remaining;
            unsigned np = pfx;
            for (int bin = 255; bin >= 0; --bin) {
                int c = (int)hist[bin];
                if (rem <= c) { np = pfx | ((unsigned)bin << shift); break; }
                rem -= c;
            }
            sh_prefix = np; sh_remaining = rem;
        }
        __syncthreads();
    }
    unsigned T = sh_prefix;
    int tieNeed = sh_remaining;
    int nG = k - tieNeed;
    if (threadIdx.x == 0) { cntG = 0; cntT = 0; }
    __syncthreads();

    int sbase = b * KTOT + c_KOFF[l];
    for (int j = threadIdx.x; j < n; j += blockDim.x) {
        float v = s[j];
        unsigned key = fkey(v);
        if (key > T) {
            int t = atomicAdd(&cntG, 1);
            candIdx[sbase + t] = c_OFF[l] + j;
            candSc[sbase + t]  = 1.0f / (1.0f + expf(-v));
        } else if (key == T) {
            int t = atomicAdd(&cntT, 1);
            if (t < tieNeed) {
                candIdx[sbase + nG + t] = c_OFF[l] + j;
                candSc[sbase + nG + t]  = 1.0f / (1.0f + expf(-v));
            }
        }
    }
}

// ---------------- kernel 3: per-image sequential NMS ----------------
__global__ void __launch_bounds__(256) nms_kernel(const float* __restrict__ prop,
                                                  const int* __restrict__ candIdx,
                                                  const float* __restrict__ candSc,
                                                  float* __restrict__ candOff,
                                                  float* __restrict__ outProps) {
    int b = blockIdx.x;
    __shared__ float s[KTOT];
    __shared__ float rv[256];
    __shared__ int   ri[256];
    __shared__ float bb[4];
    const int*   ci  = candIdx + b * KTOT;
    const float* cs  = candSc  + b * KTOT;
    float*       cob = candOff + (size_t)b * KTOT * 4;

    for (int j = threadIdx.x; j < KTOT; j += 256) {
        int gi = ci[j];
        const float* p = prop + (size_t)(b * TOT + gi) * 4;
        float x1 = p[0], y1 = p[1], x2 = p[2], y2 = p[3];
        int lvl = j / 1000; if (lvl > 4) lvl = 4;
        float off = (float)lvl * 801.0f;
        cob[j * 4 + 0] = x1 + off; cob[j * 4 + 1] = y1 + off;
        cob[j * 4 + 2] = x2 + off; cob[j * 4 + 3] = y2 + off;
        bool valid = (x2 - x1 >= 0.001f) && (y2 - y1 >= 0.001f);
        s[j] = valid ? cs[j] : -INFINITY;
    }
    __threadfence_block();
    __syncthreads();

    for (int t = 0; t < NPOST; ++t) {
        // block argmax (first index wins ties, like jnp.argmax)
        float bv = -INFINITY; int bidx = 0x7fffffff;
        for (int j = threadIdx.x; j < KTOT; j += 256) {
            float v = s[j];
            if (v > bv || (v == bv && j < bidx)) { bv = v; bidx = j; }
        }
        rv[threadIdx.x] = bv; ri[threadIdx.x] = bidx;
        __syncthreads();
        for (int stp = 128; stp > 0; stp >>= 1) {
            if (threadIdx.x < stp) {
                float v2 = rv[threadIdx.x + stp]; int i2 = ri[threadIdx.x + stp];
                if (v2 > rv[threadIdx.x] ||
                    (v2 == rv[threadIdx.x] && i2 < ri[threadIdx.x])) {
                    rv[threadIdx.x] = v2; ri[threadIdx.x] = i2;
                }
            }
            __syncthreads();
        }
        if (threadIdx.x == 0) {
            int i = ri[0];
            s[i] = -INFINITY;
            bb[0] = cob[i * 4 + 0]; bb[1] = cob[i * 4 + 1];
            bb[2] = cob[i * 4 + 2]; bb[3] = cob[i * 4 + 3];
            const float* p = prop + (size_t)(b * TOT + ci[i]) * 4;
            float* o = outProps + (size_t)(b * NPOST + t) * 4;
            o[0] = p[0]; o[1] = p[1]; o[2] = p[2]; o[3] = p[3];
        }
        __syncthreads();
        float bx1 = bb[0], by1 = bb[1], bx2 = bb[2], by2 = bb[3];
        float ba = (bx2 - bx1) * (by2 - by1);
        for (int j = threadIdx.x; j < KTOT; j += 256) {
            if (s[j] == -INFINITY) continue;
            float ox1 = cob[j*4], oy1 = cob[j*4+1], ox2 = cob[j*4+2], oy2 = cob[j*4+3];
            float a2 = (ox2 - ox1) * (oy2 - oy1);
            float ix1 = fmaxf(bx1, ox1), iy1 = fmaxf(by1, oy1);
            float ix2 = fminf(bx2, ox2), iy2 = fminf(by2, oy2);
            float iw = fmaxf(ix2 - ix1, 0.f), ih = fmaxf(iy2 - iy1, 0.f);
            float inter = iw * ih;
            float iou = inter / (ba + a2 - inter);
            if (iou > 0.7f) s[j] = -INFINITY;
        }
        __syncthreads();
    }
}

// ---------------- kernel 4: FPN level assignment ----------------
__global__ void lvl_kernel(const float* __restrict__ outProps, int* __restrict__ roiLvl) {
    int i = blockIdx.x * blockDim.x + threadIdx.x;
    if (i >= BATCH * NPOST) return;
    const float* p = outProps + (size_t)i * 4;
    float area = (p[2] - p[0]) * (p[3] - p[1]);
    float li = floorf(4.0f + log2f(sqrtf(area) / 224.0f + 1e-6f));
    li = fminf(fmaxf(li, 2.0f), 5.0f);
    roiLvl[i] = (int)li - 2;
}

// ------ kernel 5: ROI-Align: bilinear gather + pooling matmul on WMMA ------
// pooled(49 x 256) = M(49 x 196) * S(196 x 256), M has fixed 0.25 weights.
// K padded 196->224 (7 unrolled steps of 32), M padded 49->64 (4 row tiles).
// One 256-thread block per ROI = 8 wave32s = 4 rowTiles x 2 channel tiles over
// a 32-channel LDS stage (f16, K-major, stride 232 halves -> conflict-free
// ds_load_b128 B fragments). A fragments are ROI/channel-invariant (built once
// into registers); all 7 B fragments are preloaded as one ds_load clause so the
// 7 dependent WMMAs issue back-to-back with a single dscnt wait.
__global__ void __launch_bounds__(256) roialign_kernel(Ptr4 feats,
                                                       const float* __restrict__ outProps,
                                                       const int* __restrict__ roiLvl,
                                                       float* __restrict__ outFeat) {
    int blk = blockIdx.x;               // 0 .. 1999
    int b = blk / NPOST;
    int lv = roiLvl[blk];
    const float* F = feats.p[lv];
    int H = c_FH[lv], W = c_FH[lv];
    float scale = c_SC[lv];

    const float* p = outProps + (size_t)blk * 4;
    float x1 = p[0] * scale, y1 = p[1] * scale;
    float x2 = p[2] * scale, y2 = p[3] * scale;
    float rw = fmaxf(x2 - x1, 1.0f), rh = fmaxf(y2 - y1, 1.0f);

    __shared__ int      gy0[196], gx0[196], gy1[196], gx1[196];
    __shared__ float    w00[196], w01[196], w10[196], w11[196];
    __shared__ _Float16 St[32 * 232];   // [channel][K], f16, stride 232 halves

    int tid = threadIdx.x, wave = tid >> 5, lane = tid & 31;
    if (tid < 196) {
        int iy = tid / 14, ix = tid % 14;
        float ty = (iy + 0.5f) / 14.0f, tx = (ix + 0.5f) / 14.0f;
        float yy = y1 + rh * ty, xx = x1 + rw * tx;
        bool m = (yy >= -1.0f) && (yy <= (float)H) && (xx >= -1.0f) && (xx <= (float)W);
        float yc = fminf(fmaxf(yy, 0.f), (float)(H - 1));
        float xc = fminf(fmaxf(xx, 0.f), (float)(W - 1));
        int y0i = (int)floorf(yc), x0i = (int)floorf(xc);
        int y1i = y0i + 1 < H - 1 ? y0i + 1 : H - 1;
        int x1i = x0i + 1 < W - 1 ? x0i + 1 : W - 1;
        float ly = yc - (float)y0i, lx = xc - (float)x0i;
        float mm = m ? 1.0f : 0.0f;
        gy0[tid] = y0i; gx0[tid] = x0i; gy1[tid] = y1i; gx1[tid] = x1i;
        w00[tid] = (1.f - ly) * (1.f - lx) * mm;
        w01[tid] = (1.f - ly) * lx * mm;
        w10[tid] = ly * (1.f - lx) * mm;
        w11[tid] = ly * lx * mm;
    }
    // zero the K padding (196..231) for all 32 channel rows
    for (int i = tid; i < 32 * 36; i += 256) {
        int cc = i / 36, kk = 196 + i % 36;
        St[cc * 232 + kk] = (_Float16)0.f;
    }
    __syncthreads();

    int rt = wave & 3;          // pooled-row tile 0..3 (wave-uniform)
    int halfc = wave >> 2;      // 16-channel half of the 32-ch stage
    int hi = lane >> 4, l16 = lane & 15;

    // ---- A fragments (pooling weights), ROI/channel invariant: 7 x v16h ----
    v16h aFrag[7];
    {
        int prow = rt * 16 + l16;
#pragma unroll
        for (int kk = 0; kk < 7; ++kk) {
#pragma unroll
            for (int j = 0; j < 16; ++j) {
                int v4 = j >> 1, mbit = j & 1;
                int K = (v4 < 4) ? (2 * v4 + mbit + (hi ? 8 : 0))
                                 : (16 + 2 * (v4 - 4) + mbit + (hi ? 8 : 0));
                int sA = kk * 32 + K;
                int sy = sA / 14, sx = sA - sy * 14;
                int bin = (sy >> 1) * 7 + (sx >> 1);
                aFrag[kk][j] = (bin == prow) ? (_Float16)0.25f : (_Float16)0.0f;
            }
        }
    }

    size_t fbase = (size_t)b * NCH * H * W;
    int clocal = halfc * 16 + l16;

    for (int pair = 0; pair < 8; ++pair) {
        int cbase = pair * 32;
        // stage 1: bilinear gather -> f16 LDS. wave -> channel, lane -> sample
        // (lanes walk adjacent sample pixels: good feature-row locality).
        for (int u = wave; u < 32; u += 8) {
            const float* Fc = F + fbase + (size_t)(cbase + u) * H * W;
            for (int s = lane; s < 196; s += 32) {
                float v = w00[s] * Fc[gy0[s] * W + gx0[s]]
                        + w01[s] * Fc[gy0[s] * W + gx1[s]]
                        + w10[s] * Fc[gy1[s] * W + gx0[s]]
                        + w11[s] * Fc[gy1[s] * W + gx1[s]];
                St[u * 232 + s] = (_Float16)v;
            }
        }
        if (pair < 7)   // warm GL2 for next 32-channel stage -> global_prefetch_b8
            __builtin_prefetch(F + fbase + (size_t)(cbase + 32 + wave) * H * W + lane * 16, 0, 1);
        __syncthreads();

        // stage 2: pooling matmul on the matrix pipe (one job per wave).
        // Preload all 7 B fragments (one ds_load_b128 clause), then issue the
        // 7 dependent WMMAs back-to-back behind a single dscnt wait.
        const _Float16* Sb = St + clocal * 232 + hi * 8;
        v8h blo[7], bhi[7];
#pragma unroll
        for (int kk = 0; kk < 7; ++kk) {
            blo[kk] = *(const v8h*)(Sb + kk * 32);
            bhi[kk] = *(const v8h*)(Sb + kk * 32 + 16);
        }
        v8f acc = {0.f,0.f,0.f,0.f,0.f,0.f,0.f,0.f};
#pragma unroll
        for (int kk = 0; kk < 7; ++kk) {
            v16h bm = __builtin_shufflevector(blo[kk], bhi[kk],
                        0,1,2,3,4,5,6,7,8,9,10,11,12,13,14,15);
            acc = __builtin_amdgcn_wmma_f32_16x16x32_f16(
                false, aFrag[kk], false, bm, (short)0, acc, false, false);
        }

        // D layout: element v of lane -> (M = v + 8*hi, N = l16).
        // rt is wave-uniform: rt<3 rows are all valid (p = rt*16+v+8hi <= 47);
        // rt==3 keeps only p==48 (v==0, hi==0).
        int c = cbase + clocal;
        size_t obase = ((size_t)blk * NCH + c) * 49;
        if (rt < 3) {
            float* o = outFeat + obase + rt * 16 + 8 * hi;
#pragma unroll
            for (int v = 0; v < 8; ++v) o[v] = acc[v];
        } else if (hi == 0) {
            outFeat[obase + 48] = acc[0];
        }
        __syncthreads();
    }
}

// ---------------- host launcher ----------------
extern "C" void kernel_launch(void* const* d_in, const int* in_sizes, int n_in,
                              void* d_out, int out_size, void* d_ws, size_t ws_size,
                              hipStream_t stream) {
    (void)in_sizes; (void)n_in; (void)out_size; (void)ws_size;

    Ptr4 feats; Ptr5 obj, dl;
    for (int i = 0; i < 4; ++i) feats.p[i] = (const float*)d_in[1 + i];
    for (int i = 0; i < 5; ++i) obj.p[i]   = (const float*)d_in[5 + i];
    for (int i = 0; i < 5; ++i) dl.p[i]    = (const float*)d_in[10 + i];

    // workspace layout (floats)
    float* prop   = (float*)d_ws;                    // 2*TOT*4
    float* raw    = prop + (size_t)BATCH * TOT * 4;  // 2*TOT
    int*   cIdx   = (int*)(raw + (size_t)BATCH * TOT);        // 2*KTOT
    float* cSc    = (float*)(cIdx + BATCH * KTOT);            // 2*KTOT
    float* cOff   = cSc + BATCH * KTOT;                       // 2*KTOT*4
    int*   rLvl   = (int*)(cOff + (size_t)BATCH * KTOT * 4);  // 2000

    float* outProps = (float*)d_out;                 // (2,1000,4)
    float* outFeat  = outProps + BATCH * NPOST * 4;  // (2000,256,7,7)

    decode_kernel<<<(BATCH * TOT + 255) / 256, 256, 0, stream>>>(obj, dl, prop, raw);
    topk_kernel<<<BATCH * 5, 256, 0, stream>>>(raw, cIdx, cSc);
    nms_kernel<<<BATCH, 256, 0, stream>>>(prop, cIdx, cSc, cOff, outProps);
    lvl_kernel<<<(BATCH * NPOST + 255) / 256, 256, 0, stream>>>(outProps, rLvl);
    roialign_kernel<<<BATCH * NPOST, 256, 0, stream>>>(feats, outProps, rLvl, outFeat);
}
